// TreeLSTMUnit_67430986547369
// MI455X (gfx1250) — matrix operand order repaired and checked by reading, not compiled, not run on hardware
//
#include <hip/hip_runtime.h>
#include <hip/hip_bf16.h>

typedef __attribute__((ext_vector_type(16))) __bf16 v16bf;
typedef __attribute__((ext_vector_type(8)))  float  v8f;

#define BATCH 65536
#define NI 256
#define NM 256
#define NKC 40                 // 1280 / 32 K-chunks
#define ROWS_PER_BLOCK 64
#define LDS_CHUNK 49152        // 3 gates * 16 col-tiles * 1KB per K-chunk

#if defined(__AMDGCN__) && \
    __has_builtin(__builtin_amdgcn_global_load_async_to_lds_b128) && \
    __has_builtin(__builtin_amdgcn_s_wait_asynccnt)
#define USE_ASYNC_LDS 1
#else
#define USE_ASYNC_LDS 0
#endif

#if USE_ASYNC_LDS
typedef int v4i __attribute__((ext_vector_type(4)));
typedef __attribute__((address_space(1))) v4i* gv4ip_t;   // global (AS1) int4*
typedef __attribute__((address_space(3))) v4i* lv4ip_t;   // LDS (AS3) int4*
#endif

// ---------------- helpers ----------------
__device__ __forceinline__ unsigned short f2bf(float x) {
  unsigned int u = __float_as_uint(x);
  u += 0x7FFFu + ((u >> 16) & 1u);     // round-to-nearest-even
  return (unsigned short)(u >> 16);
}

#if defined(__AMDGCN__) && __has_builtin(__builtin_amdgcn_cvt_pk_bf16_f32)
typedef __attribute__((ext_vector_type(2))) __bf16 v2bf;
__device__ __forceinline__ unsigned int pack2(float a, float b) {
  union { v2bf v; unsigned int u; } cv;
  cv.v = __builtin_amdgcn_cvt_pk_bf16_f32(a, b);   // lo = a, hi = b, RNE
  return cv.u;
}
#else
__device__ __forceinline__ unsigned int pack2(float a, float b) {
  return (unsigned int)f2bf(a) | ((unsigned int)f2bf(b) << 16);
}
#endif

union Frag {
  unsigned int u[8];
  uint4 q[2];
  v16bf v;
};

// ---------------- kernel 1: pack weights into WMMA B-fragment order ----------------
// Fragment f = ((g*NKC + kc)*16 + ct); element [lane][e]:
//   K = kc*32 + (lane>=16 ? 8 : 0) + (e&7) + (e>=8 ? 16 : 0)   (ISA 16-bit B 32x16 layout)
//   N = ct*16 + (lane & 15)
__global__ void pack_weights_kernel(const float* __restrict__ Wi, const float* __restrict__ Wo,
                                    const float* __restrict__ Wu, const float* __restrict__ Ui,
                                    const float* __restrict__ Uo, const float* __restrict__ Uu,
                                    unsigned short* __restrict__ wpack) {
  int id = blockIdx.x * blockDim.x + threadIdx.x;   // 0 .. 983039
  int f      = id >> 9;
  int within = id & 511;
  int L = within >> 4;
  int e = within & 15;
  int g   = f / (NKC * 16);
  int rem = f - g * (NKC * 16);
  int kc = rem >> 4;
  int ct = rem & 15;
  int K = kc * 32 + ((L >> 4) << 3) + (e & 7) + ((e >> 3) << 4);
  int N = (ct << 4) + (L & 15);
  const float* W = (g == 0) ? Wi : (g == 1) ? Wo : Wu;
  const float* U = (g == 0) ? Ui : (g == 1) ? Uo : Uu;
  float v;
  if (K < NI) {
    v = W[N * NI + K];
  } else {
    int kk = (K - NI) >> 8;
    int m  = (K - NI) & 255;
    v = U[kk * (NM * NM) + N * NM + m];
  }
  wpack[id] = f2bf(v);
}

// ---------------- kernel 2: fused 3-gate GEMM + TreeLSTM epilogue ----------------
// block = 256 threads = 8 waves, covers 64 batch rows.
// wave w: rt-group rtg = w>>2 (2 row-tiles), col-quad ct0 = (w&3)*4.
// Per wave: 2 rt x 4 ct x 3 gates = 24 accumulator tiles.
__global__ __launch_bounds__(256)
void treelstm_wmma_kernel(const float* __restrict__ inputs,
                          const float* __restrict__ children,
                          const unsigned short* __restrict__ wpack,
                          const float* __restrict__ bi, const float* __restrict__ bo,
                          const float* __restrict__ bu,
                          float* __restrict__ out) {
#if USE_ASYNC_LDS
  extern __shared__ unsigned char smem[];    // 2 * LDS_CHUNK (double buffer)
#endif
  const int tid  = threadIdx.x;
  const int lane = tid & 31;
  const int w    = tid >> 5;
  const int rtg  = w >> 2;                   // 0..1
  const int ct0  = (w & 3) * 4;              // col-tile base
  const int rowblk = blockIdx.x * ROWS_PER_BLOCK;
  const int half = lane >> 4;

  int rbase[2], arow[2];
  rbase[0] = rowblk + rtg * 32;
  rbase[1] = rbase[0] + 16;
  arow[0] = rbase[0] + (lane & 15);
  arow[1] = rbase[1] + (lane & 15);

  v8f acc[3][4][2];
#pragma unroll
  for (int g = 0; g < 3; ++g)
#pragma unroll
    for (int j = 0; j < 4; ++j)
#pragma unroll
      for (int t = 0; t < 2; ++t)
        acc[g][j][t] = (v8f){0.f, 0.f, 0.f, 0.f, 0.f, 0.f, 0.f, 0.f};

  // per-lane activation source pointer for K-offset K, row-tile t
  auto aptr = [&](int K, int t) -> const float* {
    if (K < NI) return inputs + (size_t)arow[t] * NI + K;
    int kk = (K - NI) >> 8;
    int m  = (K - NI) & 255;
    return children + ((size_t)kk * BATCH + arow[t]) * (2 * NM) + m;
  };

#if USE_ASYNC_LDS
  // cooperative async copy of one K-chunk of packed weights (48KB) into LDS buffer
  auto issue_weights = [&](int kc, int buf) {
    const unsigned char* gsrc = (const unsigned char*)wpack;
#pragma unroll
    for (int r = 0; r < 12; ++r) {
      int c   = r * 256 + tid;              // 0..3071 16B-chunks
      int g   = c >> 10;                    // gate region
      int off = (c & 1023) * 16;
      const unsigned char* gp = gsrc + (size_t)(g * NKC + kc) * 16384 + off;
      unsigned char* lp = smem + buf * LDS_CHUNK + c * 16;
      __builtin_amdgcn_global_load_async_to_lds_b128((gv4ip_t)gp, (lv4ip_t)lp, 0, 0);
    }
  };
  issue_weights(0, 0);
#endif

  // software-pipelined raw A (f32) registers: [rt][K-group][lo/hi float4]
  float4 rawA[2][2][2];
#pragma unroll
  for (int t = 0; t < 2; ++t)
#pragma unroll
    for (int gr = 0; gr < 2; ++gr) {
      const float* p = aptr(half * 8 + gr * 16, t);
      rawA[t][gr][0] = *(const float4*)(p);
      rawA[t][gr][1] = *(const float4*)(p + 4);
    }

#if USE_ASYNC_LDS
  __builtin_amdgcn_s_wait_asynccnt(0);
  __syncthreads();
#endif

  for (int kc = 0; kc < NKC; ++kc) {
#if USE_ASYNC_LDS
    if (kc + 1 < NKC) issue_weights(kc + 1, (kc + 1) & 1);
#endif

    // ---- pack current A fragments (VALU only, inputs already resident) ----
    Frag A[2];
#pragma unroll
    for (int t = 0; t < 2; ++t)
#pragma unroll
      for (int gr = 0; gr < 2; ++gr) {
        float4 a0 = rawA[t][gr][0];
        float4 a1 = rawA[t][gr][1];
        A[t].u[gr * 4 + 0] = pack2(a0.x, a0.y);
        A[t].u[gr * 4 + 1] = pack2(a0.z, a0.w);
        A[t].u[gr * 4 + 2] = pack2(a1.x, a1.y);
        A[t].u[gr * 4 + 3] = pack2(a1.z, a1.w);
      }

    // ---- issue raw A loads for next chunk (hide HBM latency) ----
    int kcn = (kc + 1 < NKC) ? kc + 1 : kc;
    float4 rawN[2][2][2];
#pragma unroll
    for (int t = 0; t < 2; ++t)
#pragma unroll
      for (int gr = 0; gr < 2; ++gr) {
        const float* p = aptr(kcn * 32 + half * 8 + gr * 16, t);
        rawN[t][gr][0] = *(const float4*)(p);
        rawN[t][gr][1] = *(const float4*)(p + 4);
      }

    // ---- B fragments (rotating 2-deep pipeline) + 24 WMMAs ----
#if USE_ASYNC_LDS
    {
      const unsigned char* lb = smem + (kc & 1) * LDS_CHUNK + (size_t)ct0 * 1024 + lane * 32;
      Frag Bf[2];
      {
        const uint4* bp = (const uint4*)(lb);
        Bf[0].q[0] = bp[0];
        Bf[0].q[1] = bp[1];
      }
#pragma unroll
      for (int f = 0; f < 12; ++f) {
        if (f + 1 < 12) {
          int fn = f + 1;
          const uint4* bp = (const uint4*)(lb + (size_t)((fn >> 2) * 16 + (fn & 3)) * 1024);
          Bf[fn & 1].q[0] = bp[0];
          Bf[fn & 1].q[1] = bp[1];
        }
        int g = f >> 2, j = f & 3;
        acc[g][j][0] = __builtin_amdgcn_wmma_f32_16x16x32_bf16(
            false, A[0].v, false, Bf[f & 1].v, (short)0, acc[g][j][0], false, false);
        acc[g][j][1] = __builtin_amdgcn_wmma_f32_16x16x32_bf16(
            false, A[1].v, false, Bf[f & 1].v, (short)0, acc[g][j][1], false, false);
      }
    }
#else
    // fallback: direct global loads, batched 4 fragments per gate to overlap latency
#pragma unroll
    for (int g = 0; g < 3; ++g) {
      Frag Bf[4];
#pragma unroll
      for (int j = 0; j < 4; ++j) {
        const uint4* bp = (const uint4*)(wpack +
            ((size_t)((g * NKC + kc) * 16 + (ct0 + j)) * 512 + lane * 16));
        Bf[j].q[0] = bp[0];
        Bf[j].q[1] = bp[1];
      }
#pragma unroll
      for (int j = 0; j < 4; ++j)
#pragma unroll
        for (int t = 0; t < 2; ++t)
          acc[g][j][t] = __builtin_amdgcn_wmma_f32_16x16x32_bf16(
              false, A[t].v, false, Bf[j].v, (short)0, acc[g][j][t], false, false);
    }
#endif

    // rotate raw A pipeline
#pragma unroll
    for (int t = 0; t < 2; ++t)
#pragma unroll
      for (int gr = 0; gr < 2; ++gr) {
        rawA[t][gr][0] = rawN[t][gr][0];
        rawA[t][gr][1] = rawN[t][gr][1];
      }

#if USE_ASYNC_LDS
    __builtin_amdgcn_s_wait_asynccnt(0);   // chunk kc+1 landed (copied during compute)
    __syncthreads();                       // all waves done reading buffer kc&1
#endif
  }

  // ---- epilogue: bias + sigmoid/tanh, write h (cols 0..255) and c (cols 256..511) ----
  // C/D layout: N = lane&15 within col-tile; VGPR r -> M = r + 8*half.
#pragma unroll
  for (int j = 0; j < 4; ++j) {
    int col = (ct0 + j) * 16 + (lane & 15);
    float bi_v = bi[col];
    float bo_v = bo[col];
    float bu_v = bu[col];
#pragma unroll
    for (int t = 0; t < 2; ++t) {
#pragma unroll
      for (int r = 0; r < 8; ++r) {
        int row = rbase[t] + r + half * 8;
        float iv = acc[0][j][t][r] + bi_v;
        float ov = acc[1][j][t][r] + bo_v;
        float uv = acc[2][j][t][r] + bu_v;
        float sig_i = 1.f / (1.f + __expf(-iv));
        float sig_o = 1.f / (1.f + __expf(-ov));
        float cc = sig_i * tanhf(uv);
        float hh = sig_o * tanhf(cc);
        float* orow = out + (size_t)row * (2 * NM);
        orow[col]      = hh;   // h
        orow[col + NM] = cc;   // c
      }
    }
  }
}

// ---------------- launch ----------------
extern "C" void kernel_launch(void* const* d_in, const int* in_sizes, int n_in,
                              void* d_out, int out_size, void* d_ws, size_t ws_size,
                              hipStream_t stream) {
  const float* inputs   = (const float*)d_in[0];
  const float* children = (const float*)d_in[1];
  // d_in[2] arities: unused by the reference math
  const float* Wi = (const float*)d_in[3];
  const float* bi = (const float*)d_in[4];
  const float* Wo = (const float*)d_in[5];
  const float* bo = (const float*)d_in[6];
  const float* Wu = (const float*)d_in[7];
  const float* bu = (const float*)d_in[8];
  // d_in[9] Wf, d_in[10] bf, d_in[14] Uf: unused (fc_sum == 0 in reference)
  const float* Ui = (const float*)d_in[11];
  const float* Uo = (const float*)d_in[12];
  const float* Uu = (const float*)d_in[13];

  unsigned short* wpack = (unsigned short*)d_ws;  // 1.97 MB of ws
  float* out = (float*)d_out;

  pack_weights_kernel<<<983040 / 256, 256, 0, stream>>>(Wi, Wo, Wu, Ui, Uo, Uu, wpack);
  treelstm_wmma_kernel<<<BATCH / ROWS_PER_BLOCK, 256, 2 * LDS_CHUNK, stream>>>(
      inputs, children, wpack, bi, bo, bu, out);
}
